// ZoneoutLSTM_33294586479130
// MI455X (gfx1250) — matrix operand order repaired and checked by reading, not compiled, not run on hardware
//
#include <hip/hip_runtime.h>
#include <hip/hip_bf16.h>

// ---------------------------------------------------------------------------
// 2-layer LSTM (ZoneoutLSTM eval mode) for MI455X / gfx1250.
// Persistent fused kernel: per timestep, layer0 gates via bf16 WMMA, cell
// update, grid sync, layer1 gates via bf16 WMMA, cell update, grid sync.
// Weights pre-swizzled into WMMA B-operand lane layout; h state ping-pongs
// in global memory (L2-resident); c state lives in LDS.
// ---------------------------------------------------------------------------

#define T_SEQ   1024
#define BATCH   32
#define INDIM   512
#define HDIM    1024
#define NGATE   4096          // 4*H
#define NTILES  (NGATE / 16)  // 256 column tiles of 16
#define NWG     64            // persistent workgroups (each owns 16 hidden cols)

typedef __attribute__((ext_vector_type(16))) __bf16 v16bf;
typedef __attribute__((ext_vector_type(8)))  float  v8f;

struct Pack32 { uint4 lo; uint4 hi; };

// A-operand fragment: row-major [M][K] bf16 source.
// Lane L (half = L>>4, row = L&15): elems 0..7 = K k0+half*8+0..7,
// elems 8..15 = K k0+half*8+16..23  (16-bit A 16x32 layout, wave32).
__device__ __forceinline__ v16bf load_a32(const __hip_bfloat16* p) {
  Pack32 t;
  t.lo = *reinterpret_cast<const uint4*>(p);
  t.hi = *reinterpret_cast<const uint4*>(p + 16);
  return __builtin_bit_cast(v16bf, t);
}

// B-operand fragment from pre-swizzled tile: 32 lanes x 16 bf16 contiguous.
__device__ __forceinline__ v16bf load_b32(const __hip_bfloat16* p) {
  Pack32 t;
  t.lo = *reinterpret_cast<const uint4*>(p);
  t.hi = *reinterpret_cast<const uint4*>(p + 8);
  return __builtin_bit_cast(v16bf, t);
}

// One 16x16 output tile of gates = [x_t ; h] @ Wg  (K = KIN + HDIM).
// 2-deep software pipeline: loads for K-tile kt+1 in flight under WMMA kt.
template <int KIN>
__device__ __forceinline__ v8f gates_tile(const __hip_bfloat16* __restrict__ xp,
                                          const __hip_bfloat16* __restrict__ hp,
                                          const __hip_bfloat16* __restrict__ wsw,
                                          int mtile, int ntile) {
  const int lane = threadIdx.x & 31;
  const int half = lane >> 4;
  const int mr   = lane & 15;
  v8f acc = {};
  // Opaque (always-zero) SGPR element offset: defeats loop-invariant hoisting
  // of the time-invariant weight-fragment loads out of the t loop (which
  // would overflow the VGPR file and spill to scratch), while keeping the
  // pointer's global-address-space provenance so these stay global_load_b128
  // (LOADcnt only), not flat_load (LOADcnt+DScnt).
  unsigned long long wskew = 0;
  asm volatile("" : "+s"(wskew));
  const __hip_bfloat16* arow = xp + (size_t)(mtile * 16 + mr) * KIN + half * 8;
  const __hip_bfloat16* hrow = hp + (size_t)(mtile * 16 + mr) * HDIM + half * 8;
  const __hip_bfloat16* wl   = wsw + (size_t)ntile * 512 + lane * 16 + wskew;
  constexpr int KT0 = KIN / 32;
  constexpr int KT1 = HDIM / 32;
  constexpr int KT  = KT0 + KT1;

  v16bf a0 = load_a32(arow);
  v16bf b0 = load_b32(wl);
  #pragma unroll
  for (int kt = 1; kt < KT; ++kt) {
    const __hip_bfloat16* ap =
        (kt < KT0) ? (arow + kt * 32) : (hrow + (kt - KT0) * 32);
    v16bf a1 = load_a32(ap);
    v16bf b1 = load_b32(wl + (size_t)kt * (NTILES * 512));
    acc = __builtin_amdgcn_wmma_f32_16x16x32_bf16(false, a0, false, b0,
                                                  (short)0, acc, false, false);
    a0 = a1;
    b0 = b1;
  }
  acc = __builtin_amdgcn_wmma_f32_16x16x32_bf16(false, a0, false, b0,
                                                (short)0, acc, false, false);
  return acc;
}

// Device-wide barrier: monotonically increasing counter, agent scope.
__device__ __forceinline__ void grid_sync(unsigned int* bar, unsigned int& expected) {
  __threadfence();
  __syncthreads();
  if (threadIdx.x == 0) {
    __hip_atomic_fetch_add(bar, 1u, __ATOMIC_RELEASE, __HIP_MEMORY_SCOPE_AGENT);
    expected += gridDim.x;
    while (__hip_atomic_load(bar, __ATOMIC_ACQUIRE, __HIP_MEMORY_SCOPE_AGENT) < expected)
      __builtin_amdgcn_s_sleep(1);
  }
  __syncthreads();
  __threadfence();
}

// Nonlinear LSTM cell update for this WG's 16 hidden columns.
__device__ __forceinline__ void cell_update(int l, float* __restrict__ cSt,
                                            const float* __restrict__ gatesLds,
                                            const float* __restrict__ biasLds,
                                            __hip_bfloat16* __restrict__ hOut,
                                            float* __restrict__ yOut, int wg) {
  for (int e = threadIdx.x; e < BATCH * 16; e += 256) {
    const int m = e >> 4, j = e & 15;
    const float gi = gatesLds[0 * 512 + m * 16 + j] + biasLds[l * 64 + 0 * 16 + j];
    const float gf = gatesLds[1 * 512 + m * 16 + j] + biasLds[l * 64 + 1 * 16 + j];
    const float gg = gatesLds[2 * 512 + m * 16 + j] + biasLds[l * 64 + 2 * 16 + j];
    const float go = gatesLds[3 * 512 + m * 16 + j] + biasLds[l * 64 + 3 * 16 + j];
    const float ii = 1.0f / (1.0f + __expf(-gi));
    const float ff = 1.0f / (1.0f + __expf(-gf));
    const float gv = tanhf(gg);
    const float oo = 1.0f / (1.0f + __expf(-go));
    const float c  = ff * cSt[l * 512 + m * 16 + j] + ii * gv;
    cSt[l * 512 + m * 16 + j] = c;
    const float h = oo * tanhf(c);
    const int jg = wg * 16 + j;
    hOut[m * HDIM + jg] = __float2bfloat16(h);
    if (yOut) yOut[m * HDIM + jg] = h;
  }
}

__global__ __launch_bounds__(256, 1)
void zoneout_lstm_persistent(const __hip_bfloat16* __restrict__ xbf,
                             const __hip_bfloat16* __restrict__ w0sw,
                             const __hip_bfloat16* __restrict__ w1sw,
                             const float* __restrict__ bi0, const float* __restrict__ bh0,
                             const float* __restrict__ bi1, const float* __restrict__ bh1,
                             __hip_bfloat16* __restrict__ h0buf,
                             __hip_bfloat16* __restrict__ h1buf,
                             float* __restrict__ y,
                             unsigned int* __restrict__ bar) {
  __shared__ float gatesLds[4 * 32 * 16];  // [gate][m][j] pre-activations
  __shared__ float cSt[2 * 32 * 16];       // [layer][m][j] cell state
  __shared__ float biasLds[2 * 4 * 16];    // [layer][gate][j] combined bias

  const int tid = threadIdx.x;
  const int wg  = blockIdx.x;

  for (int e = tid; e < 2 * 32 * 16; e += 256) cSt[e] = 0.0f;
  if (tid < 128) {
    const int l = tid >> 6, gate = (tid >> 4) & 3, j = tid & 15;
    const int n = gate * HDIM + wg * 16 + j;
    const float* bi = l ? bi1 : bi0;
    const float* bh = l ? bh1 : bh0;
    biasLds[l * 64 + gate * 16 + j] = bi[n] + bh[n];
  }
  __syncthreads();

  const int wave  = tid >> 5;
  const int lane  = tid & 31;
  const int mtile = wave & 1;    // 2 M-tiles of 16 (B=32)
  const int gate  = wave >> 1;   // 4 gates: i,f,g,o
  const int ntile = gate * 64 + wg;
  const int half  = lane >> 4;
  const int nl    = lane & 15;

  unsigned int expected = 0;

  for (int t = 0; t < T_SEQ; ++t) {
    const int p = t & 1;
    // ---- layer 0: gates = [x_t ; h0] @ Wg0 ----
    {
      v8f acc = gates_tile<INDIM>(xbf + (size_t)t * BATCH * INDIM,
                                  h0buf + (size_t)p * BATCH * HDIM,
                                  w0sw, mtile, ntile);
      #pragma unroll
      for (int r = 0; r < 8; ++r)
        gatesLds[gate * 512 + (mtile * 16 + half * 8 + r) * 16 + nl] = acc[r];
      __syncthreads();
      cell_update(0, cSt, gatesLds, biasLds,
                  h0buf + (size_t)(1 - p) * BATCH * HDIM, nullptr, wg);
      grid_sync(bar, expected);
    }
    // ---- layer 1: gates = [h0_new ; h1] @ Wg1 ----
    {
      v8f acc = gates_tile<HDIM>(h0buf + (size_t)(1 - p) * BATCH * HDIM,
                                 h1buf + (size_t)p * BATCH * HDIM,
                                 w1sw, mtile, ntile);
      #pragma unroll
      for (int r = 0; r < 8; ++r)
        gatesLds[gate * 512 + (mtile * 16 + half * 8 + r) * 16 + nl] = acc[r];
      __syncthreads();
      cell_update(1, cSt, gatesLds, biasLds,
                  h1buf + (size_t)(1 - p) * BATCH * HDIM,
                  y + (size_t)t * BATCH * HDIM, wg);
      grid_sync(bar, expected);
    }
  }
}

// ---------------------------------------------------------------------------
// Prep kernels
// ---------------------------------------------------------------------------

__global__ void k_cvt_x(const float* __restrict__ x,
                        __hip_bfloat16* __restrict__ out, int n) {
  for (int i = blockIdx.x * blockDim.x + threadIdx.x; i < n;
       i += gridDim.x * blockDim.x)
    out[i] = __float2bfloat16(x[i]);
}

// Swizzle combined gate weights [Kin rows of W_ih^T ; H rows of W_hh^T] into
// B-operand tile layout: tile = ktile*256 + ntile, 512 bf16/tile (lane*16+e),
// element e at lane L: K = ktile*32 + (L>>4)*8 + (e&7) + (e>>3)*16, N = ntile*16 + (L&15).
__global__ void k_swizzle_w(const float* __restrict__ Wih,
                            const float* __restrict__ Whh,
                            __hip_bfloat16* __restrict__ out, int Kin, int Ktot) {
  const int total = Ktot * NGATE;
  for (int i = blockIdx.x * blockDim.x + threadIdx.x; i < total;
       i += gridDim.x * blockDim.x) {
    const int tile  = i >> 9;
    const int s     = i & 511;
    const int lane  = s >> 4;
    const int e     = s & 15;
    const int ktile = tile >> 8;   // / NTILES (=256)
    const int ntile = tile & 255;
    const int K = ktile * 32 + (lane >> 4) * 8 + (e & 7) + (e >> 3) * 16;
    const int N = ntile * 16 + (lane & 15);
    const float v = (K < Kin) ? Wih[(size_t)N * Kin + K]
                              : Whh[(size_t)N * HDIM + (K - Kin)];
    out[i] = __float2bfloat16(v);
  }
}

__global__ void k_init(unsigned int* __restrict__ hz, int nwords,
                       unsigned int* __restrict__ bar) {
  for (int i = blockIdx.x * blockDim.x + threadIdx.x; i < nwords;
       i += gridDim.x * blockDim.x)
    hz[i] = 0u;
  if (blockIdx.x == 0 && threadIdx.x == 0) *bar = 0u;
}

// ---------------------------------------------------------------------------
// Launch
// ---------------------------------------------------------------------------

extern "C" void kernel_launch(void* const* d_in, const int* in_sizes, int n_in,
                              void* d_out, int out_size, void* d_ws, size_t ws_size,
                              hipStream_t stream) {
  (void)in_sizes; (void)n_in; (void)out_size; (void)ws_size;

  const float* x    = (const float*)d_in[0];
  const float* Wih0 = (const float*)d_in[1];
  const float* Whh0 = (const float*)d_in[2];
  const float* bih0 = (const float*)d_in[3];
  const float* bhh0 = (const float*)d_in[4];
  const float* Wih1 = (const float*)d_in[5];
  const float* Whh1 = (const float*)d_in[6];
  const float* bih1 = (const float*)d_in[7];
  const float* bhh1 = (const float*)d_in[8];
  float* y = (float*)d_out;

  char* ws = (char*)d_ws;
  const size_t W0_BYTES = (size_t)(INDIM + HDIM) * NGATE * 2;  // 12 MB
  const size_t W1_BYTES = (size_t)(HDIM + HDIM) * NGATE * 2;   // 16 MB
  const size_t X_BYTES  = (size_t)T_SEQ * BATCH * INDIM * 2;   // 33.5 MB
  const size_t H_BYTES  = (size_t)2 * BATCH * HDIM * 2;        // 128 KB ping-pong

  __hip_bfloat16* w0sw = (__hip_bfloat16*)(ws);
  __hip_bfloat16* w1sw = (__hip_bfloat16*)(ws + W0_BYTES);
  __hip_bfloat16* xbf  = (__hip_bfloat16*)(ws + W0_BYTES + W1_BYTES);
  __hip_bfloat16* h0b  = (__hip_bfloat16*)(ws + W0_BYTES + W1_BYTES + X_BYTES);
  __hip_bfloat16* h1b  = (__hip_bfloat16*)(ws + W0_BYTES + W1_BYTES + X_BYTES + H_BYTES);
  unsigned int*   bar  = (unsigned int*)(ws + W0_BYTES + W1_BYTES + X_BYTES + 2 * H_BYTES);

  k_cvt_x<<<dim3(4096), dim3(256), 0, stream>>>(x, xbf, T_SEQ * BATCH * INDIM);
  k_swizzle_w<<<dim3(4096), dim3(256), 0, stream>>>(Wih0, Whh0, w0sw, INDIM, INDIM + HDIM);
  k_swizzle_w<<<dim3(4096), dim3(256), 0, stream>>>(Wih1, Whh1, w1sw, HDIM, HDIM + HDIM);
  k_init<<<dim3(64), dim3(256), 0, stream>>>((unsigned int*)h0b, (int)((2 * H_BYTES) / 4), bar);

  zoneout_lstm_persistent<<<dim3(NWG), dim3(256), 0, stream>>>(
      xbf, w0sw, w1sw, bih0, bhh0, bih1, bhh1, h0b, h1b, y, bar);
}
